// Transformer_23390391894306
// MI455X (gfx1250) — matrix operand order repaired and checked by reading, not compiled, not run on hardware
//
#include <hip/hip_runtime.h>

// ---------------------------------------------------------------------------
// CDNA5 (gfx1250) dual-tower transformer forward.
// All matmuls run on v_wmma_f32_16x16x32_bf16 (wave32), LDS-staged tiles,
// f32 accumulation; LN/softmax in f32. Layouts follow CDNA5 ISA.
// ---------------------------------------------------------------------------

typedef __bf16 bf16;
typedef __attribute__((ext_vector_type(16))) __bf16 v16bf;
typedef __attribute__((ext_vector_type(8)))  __bf16 v8bf;
typedef __attribute__((ext_vector_type(8)))  float  v8f;

static __device__ __forceinline__ bf16 tobf(float f) {
  unsigned u = __float_as_uint(f);
  unsigned r = (u + 0x7FFFu + ((u >> 16) & 1u)) >> 16;
  unsigned short s = (unsigned short)r;
  return __builtin_bit_cast(bf16, s);
}

// ---------------------------------------------------------------------------
// GEMM: C[M,N] = A[M,K] @ op(B) + bias, A/B bf16, C f32 and/or bf16.
// BNT=0: B is [K,N] row-major (normal).  BNT=1: B is [N,K] row-major (A@B^T).
// Block tile 128x64, K-step 32, 8 waves of 32x32 (2x2 WMMA tiles each).
// Batched over gridDim.z with (zb = z/nH, zh = z%nH) stride decomposition.
// Requires: M%128==0, N%64==0, K%32==0, lda/ldb/ldc multiples of 8.
// ---------------------------------------------------------------------------
template <int BNT>
__global__ __launch_bounds__(256) void gemm_bf16_kernel(
    const bf16* __restrict__ A, const bf16* __restrict__ B,
    float* __restrict__ C, bf16* __restrict__ Cb,
    const float* __restrict__ bias,
    int M, int N, int K, int lda, int ldb, int ldc,
    int nH, long long sAb, long long sAh, long long sBb, long long sBh,
    long long sCb, long long sCh, int relu) {
  __shared__ bf16 As[128 * 40];   // [128][32] padded to 40 elems/row
  __shared__ bf16 Bs[64 * 40];    // [64][32] (always stored as [N][K])

  const int tid = threadIdx.x;
  const int lane = tid & 31;
  const int wave = tid >> 5;
  const int wm = wave >> 1;       // 4 waves along M
  const int wn = wave & 1;        // 2 waves along N
  const int m0 = blockIdx.y * 128;
  const int n0 = blockIdx.x * 64;
  const long long zb = blockIdx.z / nH;
  const long long zh = blockIdx.z % nH;
  A += zb * sAb + zh * sAh;
  B += zb * sBb + zh * sBh;
  const long long co = zb * sCb + zh * sCh;

  const int half = (lane >> 4) & 1;
  const int l15 = lane & 15;

  v8f acc[2][2] = {};

  for (int k0 = 0; k0 < K; k0 += 32) {
    __syncthreads();
    // Stage A tile: 128x32 bf16 = 512 x 16B chunks (2 per thread).
#pragma unroll
    for (int c = 0; c < 2; ++c) {
      int idx = tid + c * 256;
      int r = idx >> 2, col = (idx & 3) * 8;
      *(uint4*)&As[r * 40 + col] =
          *(const uint4*)&A[(long long)(m0 + r) * lda + k0 + col];
      if (k0 + 32 < K)  // CDNA5 global_prefetch_b8 for next K tile
        __builtin_prefetch(&A[(long long)(m0 + r) * lda + k0 + 32 + col], 0, 1);
    }
    if (BNT) {  // B is [N][K]: stage rows directly.
      int r = tid >> 2, col = (tid & 3) * 8;
      *(uint4*)&Bs[r * 40 + col] =
          *(const uint4*)&B[(long long)(n0 + r) * ldb + k0 + col];
    } else {    // B is [K][N]: coalesced read, transposed LDS write.
      int kr = tid >> 3, nc = (tid & 7) * 8;
      union { uint4 u; bf16 b[8]; } tu;
      tu.u = *(const uint4*)&B[(long long)(k0 + kr) * ldb + n0 + nc];
#pragma unroll
      for (int i = 0; i < 8; ++i) Bs[(nc + i) * 40 + kr] = tu.b[i];
    }
    __syncthreads();

    // Fragment loads per ISA 16-bit A 16x32 / B 32x16 wave32 layouts.
    union Frag { v16bf v; v8bf h[2]; };
    Frag fa[2], fb[2];
#pragma unroll
    for (int i = 0; i < 2; ++i) {
      const bf16* p = &As[(wm * 32 + i * 16 + l15) * 40 + half * 8];
      fa[i].h[0] = *(const v8bf*)p;          // K = half*8 .. +7
      fa[i].h[1] = *(const v8bf*)(p + 16);   // K = half*8+16 .. +23
    }
#pragma unroll
    for (int j = 0; j < 2; ++j) {
      const bf16* p = &Bs[(wn * 32 + j * 16 + l15) * 40 + half * 8];
      fb[j].h[0] = *(const v8bf*)p;          // K = half*8 .. +7
      fb[j].h[1] = *(const v8bf*)(p + 16);   // K = half*8+16 .. +23
    }
#pragma unroll
    for (int i = 0; i < 2; ++i)
#pragma unroll
      for (int j = 0; j < 2; ++j)
        acc[i][j] = __builtin_amdgcn_wmma_f32_16x16x32_bf16(
            false, fa[i].v, false, fb[j].v, (short)0, acc[i][j], false, false);
  }

  // Epilogue: C/D layout — VGPR r: lanes 0-15 -> M=r, lanes 16-31 -> M=8+r.
#pragma unroll
  for (int i = 0; i < 2; ++i)
#pragma unroll
    for (int j = 0; j < 2; ++j) {
      union { v8f v; float f[8]; } u;
      u.v = acc[i][j];
#pragma unroll
      for (int r = 0; r < 8; ++r) {
        int mm = m0 + wm * 32 + i * 16 + half * 8 + r;
        int nn = n0 + wn * 32 + j * 16 + l15;
        float v = u.f[r];
        if (bias) v += bias[nn];
        if (relu) v = fmaxf(v, 0.f);
        long long off = co + (long long)mm * ldc + nn;
        if (C) C[off] = v;
        if (Cb) Cb[off] = tobf(v);
      }
    }
}

// ---------------------------------------------------------------------------
// Elementwise / reduction kernels
// ---------------------------------------------------------------------------
__global__ void cvt_kernel(const float* __restrict__ in, bf16* __restrict__ out,
                           long long n) {
  long long i = (long long)blockIdx.x * blockDim.x + threadIdx.x;
  if (i < n) out[i] = tobf(in[i]);
}

// xT[b,t,f] = x[b,f,t]  (f32 -> bf16)
__global__ void transpose_cvt_kernel(const float* __restrict__ x,
                                     bf16* __restrict__ xT, int F, int T) {
  long long i = (long long)blockIdx.x * blockDim.x + threadIdx.x;
  long long n = (long long)gridDim.x * blockDim.x;  // exact
  (void)n;
  int f = (int)(i % F);
  long long bt = i / F;
  int t = (int)(bt % T);
  long long b = bt / T;
  xT[i] = tobf(x[(b * F + f) * (long long)T + t]);
}

// x += sinusoidal PE; also writes bf16 copy.  n = B*T*D, exact grid.
__global__ void posenc_kernel(float* __restrict__ X, bf16* __restrict__ Xb,
                              int T, int D) {
  long long i = (long long)blockIdx.x * blockDim.x + threadIdx.x;
  int d = (int)(i % D);
  int t = (int)((i / D) % T);
  float inv = -9.210340371976184f / (float)D;  // -ln(10000)/D
  float fr = __expf((float)(d & ~1) * inv);
  float ang = (float)t * fr;
  float p = (d & 1) ? __cosf(ang) : __sinf(ang);
  float v = X[i] + p;
  X[i] = v;
  Xb[i] = tobf(v);
}

// y = LN(a + r) * g + b ; f32 + bf16 outputs. One block per row, D<=1024.
__global__ __launch_bounds__(256) void ln_kernel(
    const float* __restrict__ a, const float* __restrict__ r,
    const float* __restrict__ g, const float* __restrict__ be,
    float* __restrict__ y, bf16* __restrict__ yb, int D) {
  long long row = blockIdx.x;
  const float* pa = a + row * D;
  const float* pr = r + row * D;
  __shared__ float sb[256];
  float vv[4];
  int cnt = 0;
  float s = 0.f, sq = 0.f;
  for (int d = threadIdx.x; d < D; d += 256) {
    float v = pa[d] + pr[d];
    vv[cnt++] = v;
    s += v;
    sq += v * v;
  }
  sb[threadIdx.x] = s;
  __syncthreads();
  for (int st = 128; st > 0; st >>= 1) {
    if (threadIdx.x < st) sb[threadIdx.x] += sb[threadIdx.x + st];
    __syncthreads();
  }
  float mean = sb[0] / (float)D;
  __syncthreads();
  sb[threadIdx.x] = sq;
  __syncthreads();
  for (int st = 128; st > 0; st >>= 1) {
    if (threadIdx.x < st) sb[threadIdx.x] += sb[threadIdx.x + st];
    __syncthreads();
  }
  float var = sb[0] / (float)D - mean * mean;
  float rstd = rsqrtf(var + 1e-5f);
  cnt = 0;
  for (int d = threadIdx.x; d < D; d += 256) {
    float o = (vv[cnt++] - mean) * rstd * g[d] + be[d];
    y[row * D + d] = o;
    yb[row * D + d] = tobf(o);
  }
}

// Causal softmax on scores [Z,S,S]; one block per row; writes bf16 probs.
__global__ __launch_bounds__(256) void softmax_causal_kernel(
    const float* __restrict__ Sc, bf16* __restrict__ P, int S) {
  long long rowg = blockIdx.x;  // z*S + i
  int i = (int)(rowg % S);
  const float* src = Sc + rowg * S;
  bf16* dst = P + rowg * S;
  __shared__ float sb[256];
  float mx = -3.0e38f;
  for (int j = threadIdx.x; j <= i; j += 256) mx = fmaxf(mx, src[j]);
  sb[threadIdx.x] = mx;
  __syncthreads();
  for (int st = 128; st > 0; st >>= 1) {
    if (threadIdx.x < st)
      sb[threadIdx.x] = fmaxf(sb[threadIdx.x], sb[threadIdx.x + st]);
    __syncthreads();
  }
  float rowmax = sb[0];
  __syncthreads();
  float sum = 0.f;
  for (int j = threadIdx.x; j <= i; j += 256) sum += __expf(src[j] - rowmax);
  sb[threadIdx.x] = sum;
  __syncthreads();
  for (int st = 128; st > 0; st >>= 1) {
    if (threadIdx.x < st) sb[threadIdx.x] += sb[threadIdx.x + st];
    __syncthreads();
  }
  float inv = 1.0f / sb[0];
  for (int j = threadIdx.x; j < S; j += 256)
    dst[j] = (j <= i) ? tobf(__expf(src[j] - rowmax) * inv) : tobf(0.0f);
}

// Tall-skinny head: Ptop[b,c]=dot(Xt[b,:],W[0:K1,c]); Pbot from Xf/W[K1:,c].
__global__ __launch_bounds__(256) void head_dot_kernel(
    const float* __restrict__ Xt, const float* __restrict__ Xf,
    const float* __restrict__ W, int K1, int K2, int NC,
    float* __restrict__ Ptop, float* __restrict__ Pbot) {
  int half = blockIdx.y;
  int b = blockIdx.x / NC, c = blockIdx.x % NC;
  const float* xv;
  int K;
  long long wbase;
  if (half == 0) { xv = Xt + (long long)b * K1; K = K1; wbase = 0; }
  else           { xv = Xf + (long long)b * K2; K = K2; wbase = (long long)K1 * NC; }
  float s = 0.f;
  for (int k = threadIdx.x; k < K; k += 256)
    s += xv[k] * W[wbase + (long long)k * NC + c];
  __shared__ float sb[256];
  sb[threadIdx.x] = s;
  __syncthreads();
  for (int st = 128; st > 0; st >>= 1) {
    if (threadIdx.x < st) sb[threadIdx.x] += sb[threadIdx.x + st];
    __syncthreads();
  }
  if (threadIdx.x == 0) (half ? Pbot : Ptop)[b * NC + c] = sb[0];
}

__global__ void combine_kernel(const float* __restrict__ Gt,
                               const float* __restrict__ Gb,
                               const float* __restrict__ gate_b,
                               const float* __restrict__ Ot,
                               const float* __restrict__ Ob,
                               const float* __restrict__ out_b,
                               float* __restrict__ out, int Bsz, int NC) {
  int t = threadIdx.x;
  if (t < Bsz * NC) {
    int b = t / NC, c = t % NC;
    float z0 = Gt[b * 2 + 0] + Gb[b * 2 + 0] + gate_b[0];
    float z1 = Gt[b * 2 + 1] + Gb[b * 2 + 1] + gate_b[1];
    float m = fmaxf(z0, z1);
    float e0 = __expf(z0 - m), e1 = __expf(z1 - m);
    float g0 = e0 / (e0 + e1), g1 = e1 / (e0 + e1);
    out[t] = g0 * Ot[b * NC + c] + g1 * Ob[b * NC + c] + out_b[c];
  }
}

// ---------------------------------------------------------------------------
// Host-side tower driver (all launches on stream)
// ---------------------------------------------------------------------------
static void run_tower(hipStream_t stream, int S,
                      float* X, bf16* Xb,
                      const bf16* wq, const float* bq, const bf16* wk,
                      const float* bk, const bf16* wv, const float* bv,
                      const bf16* wo, const float* bo, const float* ln1g,
                      const float* ln1b, const bf16* w1, const float* b1,
                      const bf16* w2, const float* b2, const float* ln2g,
                      const float* ln2b,
                      bf16* Qb, bf16* Kb, bf16* Vb, float* scoreF, bf16* probB,
                      bf16* attnB, float* aF, float* fF, bf16* hB, float* x2F,
                      bf16* x2B) {
  const int Bsz = 16, D = 512, H = 8, DH = 2048, NL = 4;
  const int M = Bsz * S;
  dim3 blk(256);
  for (int l = 0; l < NL; ++l) {
    const bf16* Wq = wq + (size_t)l * D * 512;
    const bf16* Wk = wk + (size_t)l * D * 512;
    const bf16* Wv = wv + (size_t)l * D * 512;
    const bf16* Wo = wo + (size_t)l * 512 * D;
    const bf16* W1 = w1 + (size_t)l * D * DH;
    const bf16* W2 = w2 + (size_t)l * DH * D;
    const float* Bq = bq + (size_t)l * 512;
    const float* Bk = bk + (size_t)l * 512;
    const float* Bv = bv + (size_t)l * 512;
    const float* Bo = bo + (size_t)l * D;
    const float* B1 = b1 + (size_t)l * DH;
    const float* B2 = b2 + (size_t)l * D;
    const float* G1 = ln1g + (size_t)l * D;
    const float* E1 = ln1b + (size_t)l * D;
    const float* G2 = ln2g + (size_t)l * D;
    const float* E2 = ln2b + (size_t)l * D;

    // Q/K/V projections -> bf16 only.
    dim3 gp(512 / 64, M / 128, 1);
    gemm_bf16_kernel<0><<<gp, blk, 0, stream>>>(Xb, Wq, nullptr, Qb, Bq, M, 512,
                                                D, D, 512, 512, 1, 0, 0, 0, 0,
                                                0, 0, 0);
    gemm_bf16_kernel<0><<<gp, blk, 0, stream>>>(Xb, Wk, nullptr, Kb, Bk, M, 512,
                                                D, D, 512, 512, 1, 0, 0, 0, 0,
                                                0, 0, 0);
    gemm_bf16_kernel<0><<<gp, blk, 0, stream>>>(Xb, Wv, nullptr, Vb, Bv, M, 512,
                                                D, D, 512, 512, 1, 0, 0, 0, 0,
                                                0, 0, 0);
    // scores[b,h] = Q[b,:,h,:] @ K[b,:,h,:]^T  (NT), batched over B*H.
    dim3 gs(S / 64, S / 128, Bsz * H);
    gemm_bf16_kernel<1><<<gs, blk, 0, stream>>>(
        Qb, Kb, scoreF, nullptr, nullptr, S, S, 64, 512, 512, S, H,
        (long long)S * 512, 64, (long long)S * 512, 64, (long long)H * S * S,
        (long long)S * S, 0);
    // causal softmax -> bf16 probs.
    softmax_causal_kernel<<<dim3(Bsz * H * S), blk, 0, stream>>>(scoreF, probB, S);
    // attn = P @ V (NN), batched; bf16 out into head-major concat layout.
    dim3 ga(64 / 64, S / 128, Bsz * H);
    gemm_bf16_kernel<0><<<ga, blk, 0, stream>>>(
        probB, Vb, nullptr, attnB, nullptr, S, 64, S, S, 512, 512, H,
        (long long)H * S * S, (long long)S * S, (long long)S * 512, 64,
        (long long)S * 512, 64, 0);
    // out projection (f32).
    gemm_bf16_kernel<0><<<gp, blk, 0, stream>>>(attnB, Wo, aF, nullptr, Bo, M,
                                                512, 512, 512, 512, 512, 1, 0,
                                                0, 0, 0, 0, 0, 0);
    // ln1: x2 = LN(a + X)
    ln_kernel<<<dim3(M), blk, 0, stream>>>(aF, X, G1, E1, x2F, x2B, D);
    // ffn1: relu(x2 @ W1 + b1) -> bf16 only.
    dim3 gf1(DH / 64, M / 128, 1);
    gemm_bf16_kernel<0><<<gf1, blk, 0, stream>>>(x2B, W1, nullptr, hB, B1, M,
                                                 DH, D, D, DH, DH, 1, 0, 0, 0,
                                                 0, 0, 0, 1);
    // ffn2: h @ W2 + b2 -> f32.
    gemm_bf16_kernel<0><<<gp, blk, 0, stream>>>(hB, W2, fF, nullptr, B2, M, 512,
                                                DH, DH, 512, 512, 1, 0, 0, 0, 0,
                                                0, 0, 0);
    // ln2: X = LN(f + x2)
    ln_kernel<<<dim3(M), blk, 0, stream>>>(fF, x2F, G2, E2, X, Xb, D);
  }
}

// ---------------------------------------------------------------------------
extern "C" void kernel_launch(void* const* d_in, const int* in_sizes, int n_in,
                              void* d_out, int out_size, void* d_ws,
                              size_t ws_size, hipStream_t stream) {
  (void)in_sizes; (void)n_in; (void)out_size; (void)ws_size;
  const int Bsz = 16, F = 128, T = 512, D = 512, H = 8, DH = 2048, NL = 4,
            NC = 10;
  const int St = T, Sf = F;
  const int Mt = Bsz * St;  // 8192
  const int Mf = Bsz * Sf;  // 2048

  const float* x = (const float*)d_in[0];
  const float* te_W = (const float*)d_in[1];
  const float* te_b = (const float*)d_in[2];
  const float* fe_W = (const float*)d_in[3];
  const float* fe_b = (const float*)d_in[4];
  const float* gate_W = (const float*)d_in[37];
  const float* gate_b = (const float*)d_in[38];
  const float* out_W = (const float*)d_in[39];
  const float* out_b = (const float*)d_in[40];

  // Workspace bump allocator (256B aligned).
  size_t off = 0;
  auto alloc = [&](size_t bytes) -> void* {
    off = (off + 255) & ~(size_t)255;
    void* p = (char*)d_ws + off;
    off += bytes;
    return p;
  };

  // bf16 weight shadows.
  bf16* w_te = (bf16*)alloc((size_t)F * D * 2);
  bf16* w_fe = (bf16*)alloc((size_t)T * D * 2);
  bf16 *twq[2], *twk[2], *twv[2], *two[2], *tw1[2], *tw2[2];
  for (int t = 0; t < 2; ++t) {
    twq[t] = (bf16*)alloc((size_t)NL * D * 512 * 2);
    twk[t] = (bf16*)alloc((size_t)NL * D * 512 * 2);
    twv[t] = (bf16*)alloc((size_t)NL * D * 512 * 2);
    two[t] = (bf16*)alloc((size_t)NL * 512 * D * 2);
    tw1[t] = (bf16*)alloc((size_t)NL * D * DH * 2);
    tw2[t] = (bf16*)alloc((size_t)NL * DH * D * 2);
  }
  bf16* xT_b = (bf16*)alloc((size_t)Bsz * T * F * 2);
  bf16* x_b = (bf16*)alloc((size_t)Bsz * F * T * 2);
  float* Xt = (float*)alloc((size_t)Mt * D * 4);
  bf16* Xt_b = (bf16*)alloc((size_t)Mt * D * 2);
  float* Xf = (float*)alloc((size_t)Mf * D * 4);
  bf16* Xf_b = (bf16*)alloc((size_t)Mf * D * 2);
  bf16* Qb = (bf16*)alloc((size_t)Mt * 512 * 2);
  bf16* Kb = (bf16*)alloc((size_t)Mt * 512 * 2);
  bf16* Vb = (bf16*)alloc((size_t)Mt * 512 * 2);
  float* scoreF = (float*)alloc((size_t)Bsz * H * St * St * 4);  // 134 MB
  bf16* probB = (bf16*)alloc((size_t)Bsz * H * St * St * 2);     // 67 MB
  bf16* attnB = (bf16*)alloc((size_t)Mt * 512 * 2);
  float* x2F = (float*)alloc((size_t)Mt * D * 4);
  bf16* x2B = (bf16*)alloc((size_t)Mt * D * 2);
  float* Gt = (float*)alloc(Bsz * 2 * 4);
  float* Gb = (float*)alloc(Bsz * 2 * 4);
  float* Ot = (float*)alloc(Bsz * NC * 4);
  float* Ob = (float*)alloc(Bsz * NC * 4);
  // Recycle the dead score region after softmax for FFN/proj scratch.
  float* aF = scoreF;                         // Mt*D f32
  float* fF = scoreF + (size_t)Mt * D;        // Mt*D f32
  bf16* hB = (bf16*)(scoreF + 2 * (size_t)Mt * D);  // Mt*DH bf16

  dim3 blk(256);
  auto cvt = [&](const float* in, bf16* out, long long n) {
    cvt_kernel<<<dim3((unsigned)((n + 255) / 256)), blk, 0, stream>>>(in, out, n);
  };

  // Convert weights to bf16 once per call (deterministic).
  cvt(te_W, w_te, (long long)F * D);
  cvt(fe_W, w_fe, (long long)T * D);
  for (int t = 0; t < 2; ++t) {
    int base = 5 + t * 16;
    cvt((const float*)d_in[base + 0], twq[t], (long long)NL * D * 512);
    cvt((const float*)d_in[base + 2], twk[t], (long long)NL * D * 512);
    cvt((const float*)d_in[base + 4], twv[t], (long long)NL * D * 512);
    cvt((const float*)d_in[base + 6], two[t], (long long)NL * 512 * D);
    cvt((const float*)d_in[base + 10], tw1[t], (long long)NL * D * DH);
    cvt((const float*)d_in[base + 12], tw2[t], (long long)NL * DH * D);
  }
  // Tower inputs.
  transpose_cvt_kernel<<<dim3((unsigned)((size_t)Bsz * T * F / 256)), blk, 0,
                         stream>>>(x, xT_b, F, T);
  cvt(x, x_b, (long long)Bsz * F * T);

  // xt = posenc(x^T @ te_W + te_b)
  dim3 gxt(512 / 64, Mt / 128, 1);
  gemm_bf16_kernel<0><<<gxt, blk, 0, stream>>>(xT_b, w_te, Xt, nullptr, te_b,
                                               Mt, 512, F, F, 512, 512, 1, 0, 0,
                                               0, 0, 0, 0, 0);
  posenc_kernel<<<dim3((unsigned)((size_t)Mt * D / 256)), blk, 0, stream>>>(
      Xt, Xt_b, T, D);
  // xf = x @ fe_W + fe_b
  dim3 gxf(512 / 64, Mf / 128, 1);
  gemm_bf16_kernel<0><<<gxf, blk, 0, stream>>>(x_b, w_fe, Xf, nullptr, fe_b, Mf,
                                               512, T, T, 512, 512, 1, 0, 0, 0,
                                               0, 0, 0, 0);

  // Encoder towers.
  for (int t = 0; t < 2; ++t) {
    int base = 5 + t * 16;
    float* X = t == 0 ? Xt : Xf;
    bf16* Xb = t == 0 ? Xt_b : Xf_b;
    int S = t == 0 ? St : Sf;
    run_tower(stream, S, X, Xb, twq[t], (const float*)d_in[base + 1], twk[t],
              (const float*)d_in[base + 3], twv[t], (const float*)d_in[base + 5],
              two[t], (const float*)d_in[base + 7],
              (const float*)d_in[base + 8], (const float*)d_in[base + 9],
              tw1[t], (const float*)d_in[base + 11], tw2[t],
              (const float*)d_in[base + 13], (const float*)d_in[base + 14],
              (const float*)d_in[base + 15], Qb, Kb, Vb, scoreF, probB, attnB,
              aF, fF, hB, x2F, x2B);
  }

  // Gated head: split-dot avoids materializing cat/go.
  head_dot_kernel<<<dim3(Bsz * 2, 2), blk, 0, stream>>>(Xt, Xf, gate_W, T * D,
                                                        F * D, 2, Gt, Gb);
  head_dot_kernel<<<dim3(Bsz * NC, 2), blk, 0, stream>>>(Xt, Xf, out_W, T * D,
                                                         F * D, NC, Ot, Ob);
  combine_kernel<<<dim3(1), blk, 0, stream>>>(Gt, Gb, gate_b, Ot, Ob, out_b,
                                              (float*)d_out, Bsz, NC);
}
